// SA_Attention_30588757082140
// MI455X (gfx1250) — matrix operand order repaired and checked by reading, not compiled, not run on hardware
//
#include <hip/hip_runtime.h>
#include <hip/hip_bf16.h>

typedef __attribute__((ext_vector_type(16))) __bf16 v16bf;
typedef __attribute__((ext_vector_type(8)))  __bf16 v8bf;
typedef __attribute__((ext_vector_type(4)))  __bf16 v4bf;
typedef __attribute__((ext_vector_type(8)))  float  v8f;

#define SCALE 0.125f  // 64^-0.5

static __device__ __forceinline__ v16bf combine16(v8bf lo, v8bf hi) {
  v16bf r;
#pragma unroll
  for (int i = 0; i < 8; ++i) { r[i] = lo[i]; r[i + 8] = hi[i]; }
  return r;
}

// A fragment (16x32 bf16, ISA layout): lane holds row lane&15;
// K = (i&7) + (i>>3)*16 + (lane>=16 ? 8 : 0)  -> two 16B chunks per lane.
static __device__ __forceinline__ v16bf load_a_frag(const __bf16* base, int pitch_hw) {
  const int lane = threadIdx.x & 31;
  const __bf16* p = base + (lane & 15) * pitch_hw + (lane >> 4) * 8;
  return combine16(*(const v8bf*)p, *(const v8bf*)(p + 16));
}

// B fragment (32x16 bf16) read from a [N][K]-major tile (Bt):
// lane holds column lane&15, K = (i&7) + (i>>3)*16 + (lane>=16 ? 8 : 0) -> two 16B chunks per lane.
static __device__ __forceinline__ v16bf load_b_frag(const __bf16* base, int pitch_hw) {
  const int lane = threadIdx.x & 31;
  const __bf16* p = base + (lane & 15) * pitch_hw + (lane >> 4) * 8;
  return combine16(*(const v8bf*)p, *(const v8bf*)(p + 16));
}

static __device__ __forceinline__ v8f wmma_bf16(v16bf a, v16bf b, v8f c) {
  return __builtin_amdgcn_wmma_f32_16x16x32_bf16(
      /*neg_a=*/false, a, /*neg_b=*/false, b,
      /*c_mod=*/(short)0, c, /*reuse_a=*/false, /*reuse_b=*/false);
}

static __device__ __forceinline__ v8f zero8() {
  v8f z;
#pragma unroll
  for (int i = 0; i < 8; ++i) z[i] = 0.0f;
  return z;
}

static __device__ __forceinline__ v4bf cvt4(float4 v) {
  v4bf b; b[0] = (__bf16)v.x; b[1] = (__bf16)v.y;
          b[2] = (__bf16)v.z; b[3] = (__bf16)v.w;
  return b;
}

// ---------------------------------------------------------------------------
// Kernel 1: QKV projection.  C[8192 x 1536] = x[8192 x 512] @ w_qkv^T
// Double-buffered LDS; next K-slab's global loads stay in flight (distinct
// registers) while current slab's WMMAs run.  Q,K scattered head-major
// [sel][b][h][n][d]; V scattered PRE-TRANSPOSED [b][h][d][n].
// ---------------------------------------------------------------------------
__global__ __launch_bounds__(256) void qkv_gemm_kernel(
    const float* __restrict__ x, const float* __restrict__ w_qkv,
    __bf16* __restrict__ qk, __bf16* __restrict__ vt) {
  __shared__ alignas(16) __bf16 As[2][128 * 32];
  __shared__ alignas(16) __bf16 Bs[2][64 * 32];
  const int tid  = threadIdx.x;
  const int w    = tid >> 5;
  const int lane = tid & 31;
  const int hi   = lane >> 4;
  const int col  = lane & 15;
  const int bm   = blockIdx.x * 128;
  const int bn   = blockIdx.y * 64;

  v8f acc[4];
#pragma unroll
  for (int t = 0; t < 4; ++t) acc[t] = zero8();

  float4 ar[4], br[2];
  auto load_regs = [&](int kk) {
#pragma unroll
    for (int i = 0; i < 4; ++i) {
      int idx = tid + i * 256;
      ar[i] = *(const float4*)(x + (size_t)(bm + (idx >> 3)) * 512 + kk + (idx & 7) * 4);
    }
#pragma unroll
    for (int i = 0; i < 2; ++i) {
      int idx = tid + i * 256;
      br[i] = *(const float4*)(w_qkv + (size_t)(bn + (idx >> 3)) * 512 + kk + (idx & 7) * 4);
    }
  };
  auto commit = [&](int buf) {
#pragma unroll
    for (int i = 0; i < 4; ++i) {
      int idx = tid + i * 256;
      *(v4bf*)(&As[buf][(idx >> 3) * 32 + (idx & 7) * 4]) = cvt4(ar[i]);
    }
#pragma unroll
    for (int i = 0; i < 2; ++i) {
      int idx = tid + i * 256;
      *(v4bf*)(&Bs[buf][(idx >> 3) * 32 + (idx & 7) * 4]) = cvt4(br[i]);
    }
  };

  load_regs(0);
  commit(0);
  __syncthreads();

  for (int it = 0; it < 16; ++it) {
    const int cur  = it & 1;
    const bool more = (it + 1) < 16;
    if (more) load_regs((it + 1) * 32);   // loads fly under the WMMAs below

    v16bf a = load_a_frag(&As[cur][w * 16 * 32], 32);
#pragma unroll
    for (int t = 0; t < 4; ++t) {
      v16bf b = load_b_frag(&Bs[cur][t * 16 * 32], 32);
      acc[t] = wmma_bf16(a, b, acc[t]);
    }

    if (more) commit(cur ^ 1);            // waits for loads only here
    __syncthreads();
  }

#pragma unroll
  for (int t = 0; t < 4; ++t) {
    int c = bn + t * 16 + col;
    int sel = c >> 9, within = c & 511;   // sel uniform across the block-column
    int h = within >> 6, d = within & 63;
#pragma unroll
    for (int r = 0; r < 8; ++r) {
      int m = bm + w * 16 + r + hi * 8;
      int bb = m >> 11, ns = m & 2047;
      __bf16 val = (__bf16)acc[t][r];
      if (sel < 2)
        qk[((((size_t)sel * 4 + bb) * 8 + h) * 2048 + ns) * 64 + d] = val;
      else
        vt[(((size_t)bb * 8 + h) * 64 + d) * 2048 + ns] = val;   // V^T
    }
  }
}

// ---------------------------------------------------------------------------
// Kernel 2: flash attention, kv tile = 64, double-buffered K/V^T in LDS.
// Each wave owns 16 query rows; per tile: 8 WMMA (QK^T) + 8 WMMA (PV).
// ---------------------------------------------------------------------------
__global__ __launch_bounds__(256) void attn_kernel(
    const __bf16* __restrict__ qk, const __bf16* __restrict__ vt,
    __bf16* __restrict__ attn_out) {
  __shared__ alignas(16) __bf16 Ks[2][64 * 64];   // K tile, [kv][d]
  __shared__ alignas(16) __bf16 Vs[2][64 * 64];   // V^T tile, [d][kv]
  __shared__ alignas(16) __bf16 Ps[8][16 * 64];   // per-wave P staging
  const int tid  = threadIdx.x;
  const int w    = tid >> 5;
  const int lane = tid & 31;
  const int hi   = lane >> 4;
  const int col  = lane & 15;
  const int q0   = blockIdx.x * 128;
  const int h    = blockIdx.y;
  const int b    = blockIdx.z;

  const __bf16* Qp  = qk + (((size_t)0 * 4 + b) * 8 + h) * (2048 * 64);
  const __bf16* Kp  = qk + (((size_t)1 * 4 + b) * 8 + h) * (2048 * 64);
  const __bf16* Vtp = vt + ((size_t)b * 8 + h) * (size_t)(64 * 2048);

  // staging coordinates: 512 x 16B chunks per 64x64 tile, 2 per thread
  const int srow = tid >> 3;            // 0..31 (chunk 2 adds +32)
  const int sc8  = (tid & 7) * 8;

  // Q fragments (16 rows x 64 d), loaded once, straight from global
  v16bf aQ[2];
  {
    int qr = q0 + w * 16 + col;
#pragma unroll
    for (int g = 0; g < 2; ++g) {
      const __bf16* p = Qp + (size_t)qr * 64 + g * 32 + hi * 8;
      aQ[g] = combine16(*(const v8bf*)p, *(const v8bf*)(p + 16));
    }
  }

  v8f o[4];
#pragma unroll
  for (int t = 0; t < 4; ++t) o[t] = zero8();
  float mrow[8], lrow[8];
#pragma unroll
  for (int r = 0; r < 8; ++r) { mrow[r] = -__builtin_inff(); lrow[r] = 0.0f; }

  v8bf kr0, kr1, vr0, vr1;
  // prologue: stage tile 0 into buffer 0
  kr0 = *(const v8bf*)(Kp + (size_t)srow * 64 + sc8);
  kr1 = *(const v8bf*)(Kp + (size_t)(srow + 32) * 64 + sc8);
  vr0 = *(const v8bf*)(Vtp + (size_t)srow * 2048 + sc8);
  vr1 = *(const v8bf*)(Vtp + (size_t)(srow + 32) * 2048 + sc8);
  *(v8bf*)(&Ks[0][srow * 64 + sc8])        = kr0;
  *(v8bf*)(&Ks[0][(srow + 32) * 64 + sc8]) = kr1;
  *(v8bf*)(&Vs[0][srow * 64 + sc8])        = vr0;
  *(v8bf*)(&Vs[0][(srow + 32) * 64 + sc8]) = vr1;
  __syncthreads();

  for (int it = 0; it < 32; ++it) {
    const int cur  = it & 1;
    const bool more = (it + 1) < 32;
    if (more) {  // issue next tile's global loads; they fly under the WMMAs
      int kv = (it + 1) * 64;
      kr0 = *(const v8bf*)(Kp + (size_t)(kv + srow) * 64 + sc8);
      kr1 = *(const v8bf*)(Kp + (size_t)(kv + srow + 32) * 64 + sc8);
      vr0 = *(const v8bf*)(Vtp + (size_t)srow * 2048 + kv + sc8);
      vr1 = *(const v8bf*)(Vtp + (size_t)(srow + 32) * 2048 + kv + sc8);
    }

    // S = Q @ K^T  (16 x 64 block of scores)
    v8f s[4];
#pragma unroll
    for (int nt = 0; nt < 4; ++nt) s[nt] = zero8();
#pragma unroll
    for (int nt = 0; nt < 4; ++nt)
#pragma unroll
      for (int g = 0; g < 2; ++g) {
        v16bf bk = load_b_frag(&Ks[cur][nt * 16 * 64 + g * 32], 64);
        s[nt] = wmma_bf16(aQ[g], bk, s[nt]);
      }

    // online softmax per owned row (r + 8*hi); 64 columns across 16 lanes x 4
#pragma unroll
    for (int r = 0; r < 8; ++r) {
      float sv[4];
#pragma unroll
      for (int nt = 0; nt < 4; ++nt) sv[nt] = s[nt][r] * SCALE;
      float mx = fmaxf(fmaxf(sv[0], sv[1]), fmaxf(sv[2], sv[3]));
#pragma unroll
      for (int msk = 1; msk < 16; msk <<= 1)
        mx = fmaxf(mx, __shfl_xor(mx, msk, 32));
      float mnew  = fmaxf(mrow[r], mx);
      float alpha = __expf(mrow[r] - mnew);
      float rs = 0.0f;
#pragma unroll
      for (int nt = 0; nt < 4; ++nt) {
        float p = __expf(sv[nt] - mnew);
        rs += p;
        Ps[w][(r + hi * 8) * 64 + nt * 16 + col] = (__bf16)p;
      }
#pragma unroll
      for (int msk = 1; msk < 16; msk <<= 1)
        rs += __shfl_xor(rs, msk, 32);
      lrow[r] = lrow[r] * alpha + rs;
      mrow[r] = mnew;
#pragma unroll
      for (int t = 0; t < 4; ++t) o[t][r] *= alpha;
    }
    asm volatile("" ::: "memory");  // keep ds loads after ds stores (same-wave LDS is in-order)

    // O += P @ V   (P: 16x64 -> two A fragments)
    v16bf aP0 = load_a_frag(&Ps[w][0], 64);
    v16bf aP1 = load_a_frag(&Ps[w][0] + 32, 64);
#pragma unroll
    for (int t = 0; t < 4; ++t) {
      v16bf bv0 = load_b_frag(&Vs[cur][t * 16 * 64], 64);
      o[t] = wmma_bf16(aP0, bv0, o[t]);
      v16bf bv1 = load_b_frag(&Vs[cur][t * 16 * 64 + 32], 64);
      o[t] = wmma_bf16(aP1, bv1, o[t]);
    }

    if (more) {  // commit next tile into the other buffer
      const int nb = cur ^ 1;
      *(v8bf*)(&Ks[nb][srow * 64 + sc8])        = kr0;
      *(v8bf*)(&Ks[nb][(srow + 32) * 64 + sc8]) = kr1;
      *(v8bf*)(&Vs[nb][srow * 64 + sc8])        = vr0;
      *(v8bf*)(&Vs[nb][(srow + 32) * 64 + sc8]) = vr1;
    }
    __syncthreads();
  }

  // normalize and store bf16 attn output, row-major [b*2048+n][h*64+d]
#pragma unroll
  for (int r = 0; r < 8; ++r) {
    int m = q0 + w * 16 + r + hi * 8;
    float inv = 1.0f / lrow[r];
    size_t rowbase = ((size_t)b * 2048 + m) * 512 + (size_t)h * 64;
#pragma unroll
    for (int t = 0; t < 4; ++t)
      attn_out[rowbase + t * 16 + col] = (__bf16)(o[t][r] * inv);
  }
}

// ---------------------------------------------------------------------------
// Kernel 3: output projection + bias.  out = attn @ w_out^T + b_out  (f32)
// Double-buffered B tile; A fragments prefetched from global into registers.
// ---------------------------------------------------------------------------
__global__ __launch_bounds__(256) void out_gemm_kernel(
    const __bf16* __restrict__ attn, const float* __restrict__ w_out,
    const float* __restrict__ bias, float* __restrict__ out) {
  __shared__ alignas(16) __bf16 Bs[2][64 * 32];
  const int tid  = threadIdx.x;
  const int w    = tid >> 5;
  const int lane = tid & 31;
  const int hi   = lane >> 4;
  const int col  = lane & 15;
  const int bm   = blockIdx.x * 128;
  const int bn   = blockIdx.y * 64;

  v8f acc[4];
#pragma unroll
  for (int t = 0; t < 4; ++t) acc[t] = zero8();

  const __bf16* arow = attn + (size_t)(bm + w * 16 + col) * 512;

  float4 br[2];
  auto load_b_regs = [&](int kk) {
#pragma unroll
    for (int i = 0; i < 2; ++i) {
      int idx = tid + i * 256;
      br[i] = *(const float4*)(w_out + (size_t)(bn + (idx >> 3)) * 512 + kk + (idx & 7) * 4);
    }
  };
  auto commit_b = [&](int buf) {
#pragma unroll
    for (int i = 0; i < 2; ++i) {
      int idx = tid + i * 256;
      *(v4bf*)(&Bs[buf][(idx >> 3) * 32 + (idx & 7) * 4]) = cvt4(br[i]);
    }
  };

  v8bf alo, ahi2;
  load_b_regs(0);
  commit_b(0);
  alo  = *(const v8bf*)(arow + hi * 8);
  ahi2 = *(const v8bf*)(arow + hi * 8 + 16);
  __syncthreads();

  for (int it = 0; it < 16; ++it) {
    const int cur  = it & 1;
    const bool more = (it + 1) < 16;
    v8bf nlo, nhi;
    if (more) {                        // next B slab + next A fragment in flight
      load_b_regs((it + 1) * 32);
      const __bf16* p = arow + (it + 1) * 32 + hi * 8;
      nlo = *(const v8bf*)p;
      nhi = *(const v8bf*)(p + 16);
    }

    v16bf a = combine16(alo, ahi2);
#pragma unroll
    for (int t = 0; t < 4; ++t) {
      v16bf b = load_b_frag(&Bs[cur][t * 16 * 32], 32);
      acc[t] = wmma_bf16(a, b, acc[t]);
    }

    if (more) {
      commit_b(cur ^ 1);
      alo = nlo; ahi2 = nhi;
    }
    __syncthreads();
  }

#pragma unroll
  for (int t = 0; t < 4; ++t) {
    int c = bn + t * 16 + col;
    float bv = bias[c];
#pragma unroll
    for (int r = 0; r < 8; ++r)
      out[(size_t)(bm + w * 16 + r + hi * 8) * 512 + c] = acc[t][r] + bv;
  }
}

// ---------------------------------------------------------------------------
extern "C" void kernel_launch(void* const* d_in, const int* in_sizes, int n_in,
                              void* d_out, int out_size, void* d_ws, size_t ws_size,
                              hipStream_t stream) {
  (void)in_sizes; (void)n_in; (void)out_size; (void)ws_size;
  const float* x     = (const float*)d_in[0];
  const float* w_qkv = (const float*)d_in[1];
  const float* w_out = (const float*)d_in[2];
  const float* b_out = (const float*)d_in[3];
  float* out = (float*)d_out;

  __bf16* qk   = (__bf16*)d_ws;                          // Q,K: 2*4*8*2048*64 bf16 (16 MB)
  __bf16* vt   = qk + (size_t)2 * 4 * 8 * 2048 * 64;     // V^T: 4*8*64*2048 bf16   (8 MB)
  __bf16* attn = vt + (size_t)4 * 8 * 64 * 2048;         // 8192*512 bf16           (8 MB)

  qkv_gemm_kernel<<<dim3(64, 24), 256, 0, stream>>>(x, w_qkv, qk, vt);
  attn_kernel<<<dim3(16, 8, 4), 256, 0, stream>>>(qk, vt, attn);
  out_gemm_kernel<<<dim3(64, 8), 256, 0, stream>>>(attn, w_out, b_out, out);
}
